// SelfAttention_66322884984908
// MI455X (gfx1250) — matrix operand order repaired and checked by reading, not compiled, not run on hardware
//
#include <hip/hip_runtime.h>
#include <math.h>

// ---------------------------------------------------------------------------
// Problem shape (compile-time so all addressing lowers to shifts/immediates)
// ---------------------------------------------------------------------------
#define BSZ 4
#define SEQ 2048
#define DIM 1024
#define GM  (BSZ * SEQ)   // 8192 GEMM rows
#define GN  DIM
#define GK  DIM

// ---------------------------------------------------------------------------
// Types for CDNA5 WMMA (wave32): v_wmma_f32_16x16x32_bf16
// ---------------------------------------------------------------------------
typedef __attribute__((ext_vector_type(16))) __bf16 v16bf;
typedef __attribute__((ext_vector_type(8)))  __bf16 v8bf;
typedef __attribute__((ext_vector_type(8)))  float  v8f;

static __device__ __forceinline__ v8f v8f_zero() {
    v8f z;
#pragma unroll
    for (int i = 0; i < 8; ++i) z[i] = 0.0f;
    return z;
}

// Build a 16x32 bf16 A-fragment for one lane from two contiguous 8-element
// (16-byte) chunks.  Lanes 0-15 hold row M=lane with
// K = {klo..klo+15}, klo = 0 for lanes 0-15, 16 for 16-31.
static __device__ __forceinline__ v16bf make_afrag(const __bf16* p) {
    v8bf lo = *(const v8bf*)(p);
    v8bf hi = *(const v8bf*)(p + 8);
    v16bf a;
#pragma unroll
    for (int e = 0; e < 8; ++e) { a[e] = lo[e]; a[e + 8] = hi[e]; }
    return a;
}

// ---------------------------------------------------------------------------
// fp32 -> bf16 conversion
// ---------------------------------------------------------------------------
__global__ void cvt_f32_to_bf16(const float* __restrict__ in,
                                __bf16* __restrict__ out, int n) {
    int i = blockIdx.x * blockDim.x + threadIdx.x;
    if (i < n) out[i] = (__bf16)in[i];
}

// ---------------------------------------------------------------------------
// GEMM:  out[m][n] = sum_k A[m][k] * W[n][k]  (+ bias[n] when F32OUT)
// A: GM x GK bf16 row-major, W: GN x GK bf16 row-major (torch Linear weight)
// Each wave computes a 64x64 tile via 4x4 WMMA accumulators; grid exactly
// covers (GM/64)*(GN/64) = 2048 waves, 8 waves per 256-thread block.
// ---------------------------------------------------------------------------
template <bool F32OUT>
__global__ __launch_bounds__(256) void gemm_xwt_wmma_bf16(
    const __bf16* __restrict__ A, const __bf16* __restrict__ W,
    __bf16* __restrict__ outH, float* __restrict__ outF,
    const float* __restrict__ bias) {
    const int lane = threadIdx.x & 31;
    const int wid  = blockIdx.x * 8 + (threadIdx.x >> 5);
    const int tM = wid >> 4;          // GN/64 == 16 tiles per row
    const int tN = wid & 15;

    const int half = lane >> 4;
    const int l15  = lane & 15;

    const __bf16* Abase = A + (size_t)(tM * 64 + l15) * GK + half * 16;
    const __bf16* Wbase = W + (size_t)(tN * 64 + l15) * GK + half * 16;

    v8f acc[4][4];
#pragma unroll
    for (int i = 0; i < 4; ++i)
#pragma unroll
        for (int j = 0; j < 4; ++j) acc[i][j] = v8f_zero();

    for (int k0 = 0; k0 < GK; k0 += 32) {
        v16bf afrag[4], bfrag[4];
#pragma unroll
        for (int mi = 0; mi < 4; ++mi) {
            const __bf16* ap = Abase + (size_t)(mi * 16) * GK + k0;
            afrag[mi] = make_afrag(ap);
            __builtin_prefetch(ap + 32, 0, 3);   // near-scope global_prefetch
        }
#pragma unroll
        for (int ni = 0; ni < 4; ++ni) {
            // B column n over k == W row n contiguous: one 32B load
            bfrag[ni] = *(const v16bf*)(Wbase + (size_t)(ni * 16) * GK + k0);
        }
#pragma unroll
        for (int mi = 0; mi < 4; ++mi)
#pragma unroll
            for (int ni = 0; ni < 4; ++ni)
                acc[mi][ni] = __builtin_amdgcn_wmma_f32_16x16x32_bf16(
                    false, afrag[mi], false, bfrag[ni], (short)0,
                    acc[mi][ni], false, false);
    }

    // Branch-free epilogue; row stride GN between the 8 accumulator elements.
#pragma unroll
    for (int mi = 0; mi < 4; ++mi) {
        const size_t mrow = (size_t)(tM * 64 + mi * 16 + half * 8);
#pragma unroll
        for (int ni = 0; ni < 4; ++ni) {
            const int n = tN * 64 + ni * 16 + l15;
            const size_t base = mrow * GN + n;
            if (F32OUT) {
                const float bn = bias[n];
#pragma unroll
                for (int v = 0; v < 8; ++v)
                    outF[base + (size_t)v * GN] = acc[mi][ni][v] + bn;
            } else {
#pragma unroll
                for (int v = 0; v < 8; ++v)
                    outH[base + (size_t)v * GN] = (__bf16)acc[mi][ni][v];
            }
        }
    }
}

// ---------------------------------------------------------------------------
// Causal flash attention (single head, d = DIM, no 1/sqrt(d) scaling).
// Workgroup: 512 threads = 16 waves.  Each block: one (batch, 32-row q-tile).
// Wave w owns d-columns [64w, 64w+64) of O and the same d-chunk of the QK^T
// reduction; partial scores are combined across waves with ds_add_f32.
// ---------------------------------------------------------------------------
#define BQ 32
#define BK 32

__global__ __launch_bounds__(512) void attn_causal_wmma(
    const __bf16* __restrict__ Qg, const __bf16* __restrict__ Kg,
    const __bf16* __restrict__ Vg, __bf16* __restrict__ Og) {
    __shared__ alignas(32) float  s_S[BQ * BK];       // reduced scores
    __shared__ alignas(32) __bf16 s_P[BQ * BK];       // exp(S - m) bf16
    __shared__ float s_m[BQ], s_l[BQ], s_alpha[BQ];
    __shared__ alignas(32) __bf16 s_Vt[DIM * BK];     // V transposed [d][k]

    const int tid  = threadIdx.x;
    const int lane = tid & 31;
    const int w    = tid >> 5;       // 0..15: d-chunk id
    const int half = lane >> 4;
    const int l15  = lane & 15;
    const int qt    = blockIdx.x;
    const int b     = blockIdx.y;
    const int qbase = qt * BQ;

    // Per-block base pointers (batch offset folded in once)
    const __bf16* Qbase = Qg + ((size_t)b * SEQ + qbase) * DIM;
    const __bf16* Kbase = Kg + (size_t)b * SEQ * DIM;
    const __bf16* Vbase = Vg + (size_t)b * SEQ * DIM;
    __bf16*       Obase = Og + ((size_t)b * SEQ + qbase) * DIM;

    // Resident Q fragments: 2 q-subtiles x 2 d-substeps of 32
    v16bf qa[2][2];
#pragma unroll
    for (int mi = 0; mi < 2; ++mi)
#pragma unroll
        for (int ds = 0; ds < 2; ++ds)
            qa[mi][ds] = make_afrag(
                Qbase + (size_t)(mi * 16 + l15) * DIM +
                w * 64 + ds * 32 + half * 16);

    v8f oacc[2][4];
#pragma unroll
    for (int mi = 0; mi < 2; ++mi)
#pragma unroll
        for (int ni = 0; ni < 4; ++ni) oacc[mi][ni] = v8f_zero();

    if (tid < BQ) { s_m[tid] = -INFINITY; s_l[tid] = 0.0f; }
    __syncthreads();

    for (int kt = 0; kt <= qt; ++kt) {           // causal: skip upper tiles
        const int kbase = kt * BK;

        // Stage V tile transposed into LDS.  Global side is a pure linear
        // copy (DIM == 1024): vectorized 16B loads at constant 8KB strides;
        // only the LDS store address carries the transpose swizzle.
        {
            const __bf16* vsrc = Vbase + (size_t)kbase * DIM;
#pragma unroll
            for (int i = 0; i < 8; ++i) {
                const int idx = (i * 512 + tid) * 8;   // 0..32760, 8-aligned
                v8bf vv = *(const v8bf*)(vsrc + idx);
                const int kk = idx >> 10;              // key within tile
                const int dd = idx & 1023;             // d index
                __bf16* dst = &s_Vt[dd * BK + kk];
#pragma unroll
                for (int e = 0; e < 8; ++e) dst[e * BK] = vv[e];
            }
        }
        s_S[tid]       = 0.0f;
        s_S[tid + 512] = 0.0f;
        __syncthreads();

        // Partial S = Q(d-chunk) @ K(d-chunk)^T
        v8f sacc[2][2];
#pragma unroll
        for (int mi = 0; mi < 2; ++mi)
#pragma unroll
            for (int ki = 0; ki < 2; ++ki) sacc[mi][ki] = v8f_zero();
#pragma unroll
        for (int ds = 0; ds < 2; ++ds) {
            v16bf kb[2];
#pragma unroll
            for (int ki = 0; ki < 2; ++ki)
                kb[ki] = *(const v16bf*)(
                    Kbase + (size_t)(kbase + ki * 16 + l15) * DIM +
                    w * 64 + ds * 32 + half * 16);
#pragma unroll
            for (int mi = 0; mi < 2; ++mi)
#pragma unroll
                for (int ki = 0; ki < 2; ++ki)
                    sacc[mi][ki] = __builtin_amdgcn_wmma_f32_16x16x32_bf16(
                        false, qa[mi][ds], false, kb[ki], (short)0,
                        sacc[mi][ki], false, false);
        }
        // Cross-wave reduction into LDS (ds_add_f32)
#pragma unroll
        for (int mi = 0; mi < 2; ++mi)
#pragma unroll
            for (int ki = 0; ki < 2; ++ki)
#pragma unroll
                for (int v = 0; v < 8; ++v) {
                    int r = mi * 16 + half * 8 + v;
                    int c = ki * 16 + l15;
                    atomicAdd(&s_S[r * BK + c], sacc[mi][ki][v]);
                }
        __syncthreads();

        // Wave 0: online softmax update per row (causal mask inside tile)
        if (tid < BQ) {
            const int r  = tid;
            const int qg = qbase + r;
            int nvalid = qg - kbase + 1;
            if (nvalid > BK) nvalid = BK;        // >= 1 always (kt <= qt)
            float mold = s_m[r];
            float mx = mold;
            for (int c = 0; c < nvalid; ++c) mx = fmaxf(mx, s_S[r * BK + c]);
            float alpha = __expf(mold - mx);     // 0 on first tile
            float sum = 0.0f;
            for (int c = 0; c < BK; ++c) {
                float pv = (c < nvalid) ? __expf(s_S[r * BK + c] - mx) : 0.0f;
                s_P[r * BK + c] = (__bf16)pv;
                sum += pv;
            }
            s_m[r] = mx;
            s_l[r] = s_l[r] * alpha + sum;
            s_alpha[r] = alpha;
        }
        __syncthreads();

        // Rescale O accumulators by alpha(row)
#pragma unroll
        for (int mi = 0; mi < 2; ++mi) {
            float ar[8];
#pragma unroll
            for (int v = 0; v < 8; ++v)
                ar[v] = s_alpha[mi * 16 + half * 8 + v];
#pragma unroll
            for (int ni = 0; ni < 4; ++ni)
#pragma unroll
                for (int v = 0; v < 8; ++v) oacc[mi][ni][v] *= ar[v];
        }

        // O += P @ V(d-chunk): A = P from LDS, B = V^T columns (contiguous)
        v16bf pa[2];
#pragma unroll
        for (int mi = 0; mi < 2; ++mi)
            pa[mi] = make_afrag(&s_P[(mi * 16 + l15) * BK + half * 16]);
        v16bf vb[4];
#pragma unroll
        for (int ni = 0; ni < 4; ++ni)
            vb[ni] = *(const v16bf*)
                &s_Vt[(w * 64 + ni * 16 + l15) * BK + half * 16];
#pragma unroll
        for (int mi = 0; mi < 2; ++mi)
#pragma unroll
            for (int ni = 0; ni < 4; ++ni)
                oacc[mi][ni] = __builtin_amdgcn_wmma_f32_16x16x32_bf16(
                    false, pa[mi], false, vb[ni], (short)0,
                    oacc[mi][ni], false, false);
        __syncthreads();   // protect s_Vt/s_P before next-tile restage
    }

    // Epilogue: O /= l(row), store bf16
#pragma unroll
    for (int mi = 0; mi < 2; ++mi) {
        float linv[8];
#pragma unroll
        for (int v = 0; v < 8; ++v)
            linv[v] = 1.0f / s_l[mi * 16 + half * 8 + v];
#pragma unroll
        for (int ni = 0; ni < 4; ++ni) {
            const int d = w * 64 + ni * 16 + l15;
#pragma unroll
            for (int v = 0; v < 8; ++v)
                Obase[(size_t)(mi * 16 + half * 8 + v) * DIM + d] =
                    (__bf16)(oacc[mi][ni][v] * linv[v]);
        }
    }
}

// ---------------------------------------------------------------------------
// Host-side launch
// ---------------------------------------------------------------------------
extern "C" void kernel_launch(void* const* d_in, const int* in_sizes, int n_in,
                              void* d_out, int out_size, void* d_ws,
                              size_t ws_size, hipStream_t stream) {
    (void)in_sizes; (void)n_in; (void)out_size; (void)ws_size;
    const size_t MS = (size_t)GM;

    const float* X  = (const float*)d_in[0];
    const float* WQ = (const float*)d_in[1];
    const float* WK = (const float*)d_in[2];
    const float* WV = (const float*)d_in[3];
    const float* WO = (const float*)d_in[4];
    const float* bO = (const float*)d_in[5];

    char* p = (char*)d_ws;
    __bf16* Xb  = (__bf16*)p; p += MS * DIM * 2;
    __bf16* WQb = (__bf16*)p; p += (size_t)DIM * DIM * 2;
    __bf16* WKb = (__bf16*)p; p += (size_t)DIM * DIM * 2;
    __bf16* WVb = (__bf16*)p; p += (size_t)DIM * DIM * 2;
    __bf16* WOb = (__bf16*)p; p += (size_t)DIM * DIM * 2;
    __bf16* Qb  = (__bf16*)p; p += MS * DIM * 2;
    __bf16* Kb  = (__bf16*)p; p += MS * DIM * 2;
    __bf16* Vb  = (__bf16*)p; p += MS * DIM * 2;
    __bf16* Ob  = (__bf16*)p; p += MS * DIM * 2;

    // fp32 -> bf16
    {
        int n = (int)(MS * DIM);
        cvt_f32_to_bf16<<<(n + 255) / 256, 256, 0, stream>>>(X, Xb, n);
        int nw = DIM * DIM;
        cvt_f32_to_bf16<<<(nw + 255) / 256, 256, 0, stream>>>(WQ, WQb, nw);
        cvt_f32_to_bf16<<<(nw + 255) / 256, 256, 0, stream>>>(WK, WKb, nw);
        cvt_f32_to_bf16<<<(nw + 255) / 256, 256, 0, stream>>>(WV, WVb, nw);
        cvt_f32_to_bf16<<<(nw + 255) / 256, 256, 0, stream>>>(WO, WOb, nw);
    }

    // QKV projections: (8192 x 1024) @ (1024 x 1024)^T
    const int gemmBlocks = (GM / 64) * (GN / 64) / 8;   // 256 blocks
    gemm_xwt_wmma_bf16<false><<<gemmBlocks, 256, 0, stream>>>(
        Xb, WQb, Qb, nullptr, nullptr);
    gemm_xwt_wmma_bf16<false><<<gemmBlocks, 256, 0, stream>>>(
        Xb, WKb, Kb, nullptr, nullptr);
    gemm_xwt_wmma_bf16<false><<<gemmBlocks, 256, 0, stream>>>(
        Xb, WVb, Vb, nullptr, nullptr);

    // Causal attention
    attn_causal_wmma<<<dim3(SEQ / BQ, BSZ), 512, 0, stream>>>(
        Qb, Kb, Vb, Ob);

    // Output projection (fp32 out + bias)
    gemm_xwt_wmma_bf16<true><<<gemmBlocks, 256, 0, stream>>>(
        Ob, WOb, nullptr, (float*)d_out, bO);
}
